// PositizedConv2d_49014166782408
// MI455X (gfx1250) — compile-verified, matrix operand order checked by reading, not run on hardware
//
#include <hip/hip_runtime.h>
#include <hip/hip_bf16.h>

typedef __attribute__((ext_vector_type(16))) _Float16 v16h;
typedef __attribute__((ext_vector_type(8)))  _Float16 v8h;
typedef __attribute__((ext_vector_type(8)))  float    v8f;

// ---- posit(8,1) round-to-nearest, exact in fp16 (<=4 frac bits, exp in [-12,13])
__device__ __forceinline__ _Float16 posit8_1_h(float x) {
    const float maxpos = 4096.0f;          // 2^(2*6)
    const float minpos = 1.0f / 4096.0f;   // 2^-12
    float ax = fabsf(x);
    ax = fminf(fmaxf(ax, minpos), maxpos);
    unsigned u = __float_as_uint(ax);
    int e = (int)((u >> 23) & 0xFF) - 127;       // floor(log2(ax)), ax normal
    int k = e >> 1;                              // floor(e/2), arithmetic shift
    int rlen = (k >= 0) ? (k + 2) : (1 - k);     // regime length
    int fb = 6 - rlen; fb = fb < 0 ? 0 : fb;     // surviving fraction bits (<=4)
    float m     = __uint_as_float((u & 0x007FFFFFu) | 0x3F800000u);   // mantissa in [1,2)
    float step  = __uint_as_float((unsigned)(127 + fb) << 23);        // 2^fb
    float rstep = __uint_as_float((unsigned)(127 - fb) << 23);        // 2^-fb
    float frac  = rintf((m - 1.0f) * step) * rstep;                   // RNE, matches jnp.round
    float scale = __uint_as_float((unsigned)(e + 127) << 23);         // 2^e
    float q = (1.0f + frac) * scale;
    q = fminf(fmaxf(q, minpos), maxpos);
    q = copysignf(q, x);
    return (_Float16)((x == 0.0f) ? 0.0f : q);
}

#define HT 8     // output rows per block
#define WT 64    // output cols per block
#define XROWS (HT + 2)
#define XCOLS (WT + 2)

__global__ __launch_bounds__(256)
void positized_conv3x3_wmma(const float* __restrict__ xin,
                            const float* __restrict__ wgt,
                            const float* __restrict__ bias,
                            float* __restrict__ out) {
    // LDS: quantized input tile [y][x][c] (channels contiguous -> ds_load_b128),
    //      quantized weights [tap][o][c] with tap 9 zeroed,
    //      per-wave C staging for coalesced output.
    __shared__ __align__(16) _Float16 xs[XROWS * XCOLS * 16];   // 21120 B
    __shared__ __align__(16) _Float16 wq[10 * 16 * 16];         //  5120 B
    __shared__ __align__(16) float    cstage[8][16 * 16];       //  8192 B

    const int n  = blockIdx.z;
    const int h0 = blockIdx.y * HT;
    const int x0 = blockIdx.x * WT;

    // ---- stage 1: global -> quantize -> LDS (coalesced along W)
    const int TILE_ELEMS = XROWS * XCOLS * 16;   // 10560
    for (int idx = threadIdx.x; idx < TILE_ELEMS; idx += 256) {
        int col = idx % XCOLS;
        int t   = idx / XCOLS;
        int c   = t % 16;
        int row = t / 16;
        int gy = h0 - 1 + row;
        int gx = x0 - 1 + col;
        _Float16 hv = (_Float16)0.0f;
        if ((unsigned)gy < 256u && (unsigned)gx < 256u) {
            float v = xin[(((size_t)n * 16 + c) * 256 + gy) * 256 + gx];
            hv = posit8_1_h(v);
        }
        xs[(row * XCOLS + col) * 16 + c] = hv;
    }
    // weights: 10 taps x 16 o x 16 c = 2560 (tap 9 = zero pad for the K=144->160 chunk)
    for (int idx = threadIdx.x; idx < 2560; idx += 256) {
        int c = idx & 15;
        int o = (idx >> 4) & 15;
        int t = idx >> 8;
        _Float16 hv = (_Float16)0.0f;
        if (t < 9) {
            int ky = t / 3, kx = t % 3;
            hv = posit8_1_h(wgt[((o * 16 + c) * 3 + ky) * 3 + kx]);
        }
        wq[(t * 16 + o) * 16 + c] = hv;
    }
    __syncthreads();

    // ---- stage 2: implicit GEMM via v_wmma_f32_16x16x32_f16
    const int wave = threadIdx.x >> 5;   // 8 waves, one output row each
    const int lane = threadIdx.x & 31;
    const int Mi   = lane & 15;          // A-matrix row = pixel within 16-wide tile / B col = o
    const int hs   = lane >> 4;          // half-wave select
    const int cb   = hs * 8;             // channel base for this half-wave's K stripe

    // B fragments are invariant: 5 K-chunks of (tapA, tapB) channels
    v16h Bf[5];
    {
        const int o = Mi;
        v8h z = {};
#pragma unroll
        for (int ch = 0; ch < 5; ++ch) {
            int tA = 2 * ch, tB = 2 * ch + 1;   // tB==9 slot is zero-filled in wq
            v8h blo = *(const v8h*)&wq[(tA * 16 + o) * 16 + cb];
            v8h bhi = *(const v8h*)&wq[(tB * 16 + o) * 16 + cb];
            (void)z;
            Bf[ch] = __builtin_shufflevector(blo, bhi,
                                             0, 1, 2, 3, 4, 5, 6, 7,
                                             8, 9, 10, 11, 12, 13, 14, 15);
        }
    }
    const float bv = bias[Mi];           // bias for output channel o = Mi

    for (int q = 0; q < 4; ++q) {
        const int wl = q * 16 + Mi;      // interior col of this pixel; +dx gives padded coord
        v8f acc;
#pragma unroll
        for (int r = 0; r < 8; ++r) acc[r] = bv;

#pragma unroll
        for (int ch = 0; ch < 4; ++ch) {
            int tA = 2 * ch, tB = tA + 1;
            int yA = wave + tA / 3, xA = wl + tA % 3;
            int yB = wave + tB / 3, xB = wl + tB % 3;
            v8h alo = *(const v8h*)&xs[(yA * XCOLS + xA) * 16 + cb];
            v8h ahi = *(const v8h*)&xs[(yB * XCOLS + xB) * 16 + cb];
            v16h A = __builtin_shufflevector(alo, ahi,
                                             0, 1, 2, 3, 4, 5, 6, 7,
                                             8, 9, 10, 11, 12, 13, 14, 15);
            acc = __builtin_amdgcn_wmma_f32_16x16x32_f16(
                      false, A, false, Bf[ch], (short)0, acc, false, false);
        }
        {   // chunk 4: tap 8 (dy=2,dx=2) + zero-padded tap
            int yA = wave + 2, xA = wl + 2;
            v8h alo = *(const v8h*)&xs[(yA * XCOLS + xA) * 16 + cb];
            v8h z = {};
            v16h A = __builtin_shufflevector(alo, z,
                                             0, 1, 2, 3, 4, 5, 6, 7,
                                             8, 9, 10, 11, 12, 13, 14, 15);
            acc = __builtin_amdgcn_wmma_f32_16x16x32_f16(
                      false, A, false, Bf[4], (short)0, acc, false, false);
        }

        // ---- stage 3: transpose C through wave-private LDS for coalesced stores.
        // C layout: acc[r] = C[M = r + 8*hs][N = Mi]; stage as cstage[N][M].
        float* cs = &cstage[wave][0];
#pragma unroll
        for (int r = 0; r < 8; ++r)
            cs[Mi * 16 + (r + 8 * hs)] = acc[r];
        asm volatile("s_wait_dscnt 0" ::: "memory");
        __builtin_amdgcn_wave_barrier();

        const int h   = h0 + wave;
        const int w0g = x0 + q * 16;
#pragma unroll
        for (int r = 0; r < 8; ++r) {
            int oo = 2 * r + hs;   // 16 lanes -> 16 contiguous floats per channel row
            out[(((size_t)n * 16 + oo) * 256 + h) * 256 + w0g + Mi] = cs[oo * 16 + Mi];
        }
        asm volatile("s_wait_dscnt 0" ::: "memory");   // WAR guard before next q reuses cs
        __builtin_amdgcn_wave_barrier();
    }
}

extern "C" void kernel_launch(void* const* d_in, const int* in_sizes, int n_in,
                              void* d_out, int out_size, void* d_ws, size_t ws_size,
                              hipStream_t stream) {
    const float* x = (const float*)d_in[0];   // (32,16,256,256)
    const float* W = (const float*)d_in[1];   // (16,16,3,3)
    const float* b = (const float*)d_in[2];   // (16,)
    float* out = (float*)d_out;               // (32,16,256,256)
    dim3 grid(256 / WT, 256 / HT, 32);        // (4, 32, 32)
    dim3 block(256, 1, 1);
    positized_conv3x3_wmma<<<grid, block, 0, stream>>>(x, W, b, out);
}
